// MSA_32796370272773
// MI455X (gfx1250) — compile-verified, matrix operand order checked by reading, not run on hardware
//
#include <hip/hip_runtime.h>
#include <cmath>

// ---------------------------------------------------------------------------
// MSA layer for MI455X (gfx1250, wave32, WMMA, async global->LDS).
// All large GEMMs: v_wmma_f32_16x16x32_bf16, bf16-resident operands, LDS
// double-buffered 256x64 tiles (64x32 per wave => 8 WMMA per 6 frag loads),
// async tile staging (ASYNCcnt) via global_load_async_to_lds_b128.
// ---------------------------------------------------------------------------

typedef __attribute__((ext_vector_type(8)))  float          v8f;
typedef __attribute__((ext_vector_type(8)))  unsigned short u16x8;
typedef __attribute__((ext_vector_type(16))) unsigned short u16x16;
typedef __attribute__((ext_vector_type(16))) __bf16         bf16x16;
typedef unsigned short ushort_t;

// exact pointee type the async-LDS builtin expects (per hipcc diagnostic)
typedef __attribute__((vector_size(16))) int v4i_t;
typedef __attribute__((address_space(1))) v4i_t* as1_v4i_p;
typedef __attribute__((address_space(3))) v4i_t* as3_v4i_p;

#define BM 256
#define BN 64
#define BK 32
#define LDS_STRIDE 40   // 32 + 8 pad: 16B-aligned rows, conflict-free frags

#if __has_builtin(__builtin_amdgcn_global_load_async_to_lds_b128)
#define ASYNC_LDS_COPY 1
#else
#define ASYNC_LDS_COPY 0
#endif

__device__ __forceinline__ ushort_t f2bf(float f) {
  unsigned u = __builtin_bit_cast(unsigned, f);
  u += 0x7FFFu + ((u >> 16) & 1u);      // round-to-nearest-even
  return (ushort_t)(u >> 16);
}
__device__ __forceinline__ float bf2f(ushort_t u) {
  return __builtin_bit_cast(float, (unsigned)u << 16);
}
__device__ __forceinline__ float gelu_exact(float x) {
  return 0.5f * x * (1.0f + erff(x * 0.70710678118654752f));
}

// 16B global(bf16) -> LDS copy; async on CDNA5 when available.
// Integer-cast route to the AS(1)/AS(3) typed pointers (AS3 ptrs are 32-bit,
// so the int->ptr conversion truncates the generic LDS address to its offset).
__device__ __forceinline__ void cp_b128(const ushort_t* g, ushort_t* l) {
#if ASYNC_LDS_COPY
  __builtin_amdgcn_global_load_async_to_lds_b128(
      (as1_v4i_p)(unsigned long long)g,
      (as3_v4i_p)(unsigned long long)l, 0, 0);
#else
  *(u16x8*)l = *(const u16x8*)g;
#endif
}
__device__ __forceinline__ void async_wait0() {
#if ASYNC_LDS_COPY
#if __has_builtin(__builtin_amdgcn_s_wait_asynccnt)
  __builtin_amdgcn_s_wait_asynccnt(0);
#else
  asm volatile("s_wait_asynccnt 0" ::: "memory");
#endif
#endif
}

// Fragment load matching CDNA5 16-bit A/B VGPR layout:
// lanes 0-15: K 0-7 (elems 0-7), K 16-23 (elems 8-15); lanes 16-31: +8.
__device__ __forceinline__ bf16x16 load_frag(const ushort_t* tileRow0, int lane) {
  const int m  = lane & 15;
  const int kh = (lane >> 4) << 3;      // 0 or 8
  const ushort_t* p = tileRow0 + m * LDS_STRIDE + kh;
  u16x8 lo = *(const u16x8*)(p);        // K = kh .. kh+7
  u16x8 hi = *(const u16x8*)(p + 16);   // K = kh+16 .. kh+23
  u16x16 r;
#pragma unroll
  for (int i = 0; i < 8; ++i) { r[i] = lo[i]; r[i + 8] = hi[i]; }
  return __builtin_bit_cast(bf16x16, r);
}

// Y[M,N] = act(alpha * A @ B^T + bias); A,B bf16; C dual fp32/bf16 (nullable).
__global__ __launch_bounds__(256) void gemm_wmma_kernel(
    const ushort_t* __restrict__ A, long lda,
    const ushort_t* __restrict__ B, long ldb, int bKmajor,
    const float* __restrict__ bias,
    float* __restrict__ C32, ushort_t* __restrict__ C16, long ldc,
    int M, int N, int K, float alpha, int act,
    int innerB, long sAo, long sAi, long sBo, long sBi, long sCo, long sCi)
{
  __shared__ __align__(16) ushort_t As[2][BM][LDS_STRIDE];
  __shared__ __align__(16) ushort_t Bs[2][BN][LDS_STRIDE];

  const int tid   = threadIdx.x;
  const int lane  = tid & 31;
  const int w     = tid >> 5;
  const int waveM = w >> 1;             // 0..3  (64 rows each)
  const int waveN = w & 1;              // 0..1  (32 cols each)

  const int z  = blockIdx.z;
  const int zo = z / innerB, zi = z % innerB;
  A += (long)zo * sAo + (long)zi * sAi;
  B += (long)zo * sBo + (long)zi * sBi;
  const long cOff = (long)zo * sCo + (long)zi * sCi;

  const int m0 = blockIdx.y * BM;
  const int n0 = blockIdx.x * BN;

  const v8f vzero = {0.f, 0.f, 0.f, 0.f, 0.f, 0.f, 0.f, 0.f};
  v8f acc[4][2];
#pragma unroll
  for (int i = 0; i < 4; ++i)
#pragma unroll
    for (int j = 0; j < 2; ++j) acc[i][j] = vzero;

  const int KT = K / BK;

  auto stage = [&](int kt, int b) {
    const int k0 = kt * BK;
#pragma unroll
    for (int p = 0; p < 4; ++p) {                 // A: 256x32 = 1024 16B chunks
      const int chunk = p * 256 + tid;
      const int r = chunk >> 2;
      const int c = (chunk & 3) * 8;
      cp_b128(&A[(long)(m0 + r) * lda + k0 + c], &As[b][r][c]);
    }
    if (!bKmajor) {                               // B: 64x32 [N,K] row-major
      const int r = tid >> 2;
      const int c = (tid & 3) * 8;
      cp_b128(&B[(long)(n0 + r) * ldb + k0 + c], &Bs[b][r][c]);
    } else {                                      // B: [K,N] -> transpose stage
      const int k  = tid >> 3;
      const int nb = (tid & 7) * 8;
      u16x8 v = *(const u16x8*)&B[(long)(k0 + k) * ldb + n0 + nb];
#pragma unroll
      for (int i = 0; i < 8; ++i) Bs[b][nb + i][k] = v[i];
    }
  };

  stage(0, 0);
  async_wait0();
  __syncthreads();

  for (int kt = 0; kt < KT; ++kt) {
    const int cur = kt & 1;
    if (kt + 1 < KT) stage(kt + 1, cur ^ 1);

    bf16x16 aF[4], bF[2];
#pragma unroll
    for (int mt = 0; mt < 4; ++mt)
      aF[mt] = load_frag(&As[cur][waveM * 64 + mt * 16][0], lane);
#pragma unroll
    for (int nt = 0; nt < 2; ++nt)
      bF[nt] = load_frag(&Bs[cur][waveN * 32 + nt * 16][0], lane);
#pragma unroll
    for (int mt = 0; mt < 4; ++mt)
#pragma unroll
      for (int nt = 0; nt < 2; ++nt)
        acc[mt][nt] = __builtin_amdgcn_wmma_f32_16x16x32_bf16(
            false, aF[mt], false, bF[nt], (short)0, acc[mt][nt], false, false);

    async_wait0();
    __syncthreads();
  }

  // C/D layout: VGPR r -> row = r + (lane>=16 ? 8 : 0); col = lane & 15
  const int rbase = (lane >> 4) * 8;
  const int cl    = lane & 15;
#pragma unroll
  for (int mt = 0; mt < 4; ++mt) {
#pragma unroll
    for (int nt = 0; nt < 2; ++nt) {
      const int gmB = m0 + waveM * 64 + mt * 16 + rbase;
      const int gn  = n0 + waveN * 32 + nt * 16 + cl;
      const float bv = bias ? bias[gn] : 0.f;
#pragma unroll
      for (int r = 0; r < 8; ++r) {
        float x = acc[mt][nt][r] * alpha + bv;
        if (act == 1) x = gelu_exact(x);
        const long ci = cOff + (long)(gmB + r) * ldc + gn;
        if (C32) C32[ci] = x;
        if (C16) C16[ci] = f2bf(x);
      }
    }
  }
}

// ---------------------------------------------------------------------------
__global__ __launch_bounds__(256) void f32_to_bf16_kernel(
    const float* __restrict__ src, ushort_t* __restrict__ dst, long n)
{
  const long i = ((long)blockIdx.x * 256 + threadIdx.x) * 4;
  if (i + 3 < n) {
    float4 v = *(const float4*)(src + i);
    dst[i + 0] = f2bf(v.x); dst[i + 1] = f2bf(v.y);
    dst[i + 2] = f2bf(v.z); dst[i + 3] = f2bf(v.w);
  }
}

// Label attention: scores0 (raw, tuple output), softmax over L=50, first=A@E,
// h_truth = lhs + first (bf16), h_unk = lhs + tp (bf16). One block per row.
__global__ __launch_bounds__(256) void label_attn_kernel(
    const float* __restrict__ lhs, const float* __restrict__ emb,
    const float* __restrict__ tp, float* __restrict__ scores0,
    float* __restrict__ first, ushort_t* __restrict__ h_unk,
    ushort_t* __restrict__ h_truth)
{
  __shared__ float row[768];
  __shared__ float sc[64];
  const long r = blockIdx.x;
  const int tid = threadIdx.x;
  const float* x = lhs + r * 768;
  for (int i = tid; i < 768; i += 256) row[i] = x[i];
  __syncthreads();

  if (tid < 50) {
    const float* e = emb + (long)tid * 768;
    float s = 0.f;
    for (int i = 0; i < 768; ++i) s = fmaf(row[i], e[i], s);
    sc[tid] = s;
    scores0[r * 50 + tid] = s;
  }
  __syncthreads();

  if (tid < 32) {  // wave 0: softmax over 50 labels
    float v0 = (tid      < 50) ? sc[tid]      : -INFINITY;
    float v1 = (tid + 32 < 50) ? sc[tid + 32] : -INFINITY;
    float m = fmaxf(v0, v1);
#pragma unroll
    for (int off = 16; off; off >>= 1) m = fmaxf(m, __shfl_xor(m, off, 32));
    float e0 = (tid      < 50) ? __expf(v0 - m) : 0.f;
    float e1 = (tid + 32 < 50) ? __expf(v1 - m) : 0.f;
    float s = e0 + e1;
#pragma unroll
    for (int off = 16; off; off >>= 1) s += __shfl_xor(s, off, 32);
    const float inv = 1.f / s;
    if (tid      < 50) sc[tid]      = e0 * inv;
    if (tid + 32 < 50) sc[tid + 32] = e1 * inv;
  }
  __syncthreads();

  for (int h = tid; h < 768; h += 256) {
    float f = 0.f;
#pragma unroll 5
    for (int l = 0; l < 50; ++l) f = fmaf(sc[l], emb[l * 768 + h], f);
    const float lv = row[h];
    first[r * 768 + h]   = f;
    h_truth[r * 768 + h] = f2bf(lv + f);
    h_unk[r * 768 + h]   = f2bf(lv + tp[h]);
  }
}

// s2[b,h,t] = q_u . k_u over the 64-wide head slice (bf16 operands)
__global__ __launch_bounds__(256) void s2_diag_kernel(
    const ushort_t* __restrict__ qu, const ushort_t* __restrict__ ku,
    float* __restrict__ s2)
{
  const int idx = blockIdx.x * 256 + threadIdx.x;   // < 96*512
  const int zz = idx >> 9, t = idx & 511;
  const int b = zz / 12, head = zz % 12;
  const long off = (long)(b * 512 + t) * 768 + head * 64;
  const ushort_t* q = qu + off;
  const ushort_t* k = ku + off;
  float s = 0.f;
#pragma unroll
  for (int i = 0; i < 64; ++i) s = fmaf(bf2f(q[i]), bf2f(k[i]), s);
  s2[idx] = s;
}

// Row softmax over 512 with diag substitution + mask; emits bf16 P and pdiag.
__global__ __launch_bounds__(256) void attn_softmax_kernel(
    const float* __restrict__ scores, const float* __restrict__ s2,
    const int* __restrict__ amask, ushort_t* __restrict__ Pbf,
    float* __restrict__ pdiag, float alpha)
{
  const int rr   = blockIdx.x * 8 + (threadIdx.x >> 5);
  const int lane = threadIdx.x & 31;
  const int zz = rr >> 9, tq = rr & 511;
  const int b = zz / 12;
  const float* row = scores + (long)rr * 512;
  ushort_t* rowo = Pbf + (long)rr * 512;
  const int* mrow = amask + b * 512;

  float vals[16];
  float mx = -INFINITY;
#pragma unroll
  for (int i = 0; i < 16; ++i) {
    const int j = lane + (i << 5);
    float v = (j == tq) ? s2[rr] * alpha : row[j];
    v += (1.0f - (float)mrow[j]) * -1e9f;
    vals[i] = v;
    mx = fmaxf(mx, v);
  }
#pragma unroll
  for (int off = 16; off; off >>= 1) mx = fmaxf(mx, __shfl_xor(mx, off, 32));
  float sum = 0.f;
#pragma unroll
  for (int i = 0; i < 16; ++i) { vals[i] = __expf(vals[i] - mx); sum += vals[i]; }
#pragma unroll
  for (int off = 16; off; off >>= 1) sum += __shfl_xor(sum, off, 32);
  const float inv = 1.f / sum;
#pragma unroll
  for (int i = 0; i < 16; ++i) {
    const int j = lane + (i << 5);
    const float p = vals[i] * inv;
    rowo[j] = f2bf(p);
    if (j == tq) pdiag[rr] = p;
  }
}

// ctx += pdiag * (v_u - v_t); dual-store fp32 + bf16 (A of the Wso GEMM)
__global__ __launch_bounds__(256) void ctx_diag_fix_kernel(
    float* __restrict__ ctx, ushort_t* __restrict__ ctxbf,
    const float* __restrict__ pdiag,
    const float* __restrict__ vu, const float* __restrict__ vt)
{
  const long idx = (long)blockIdx.x * 256 + threadIdx.x;
  const int row = (int)(idx / 768), col = (int)(idx % 768);
  const int head = col >> 6;
  const int b = row >> 9, t = row & 511;
  const float pd = pdiag[((long)(b * 12 + head) << 9) + t];
  const float v = ctx[idx] + pd * (vu[idx] - vt[idx]);
  ctx[idx] = v;
  ctxbf[idx] = f2bf(v);
}

// out = LN(a [+ badd]) * g + beta [+ r1] [+ r2]; optional bf16 mirror
__global__ __launch_bounds__(256) void layernorm_kernel(
    const float* __restrict__ a, const float* __restrict__ badd,
    const float* __restrict__ g, const float* __restrict__ beta,
    const float* __restrict__ r1, const float* __restrict__ r2,
    float* __restrict__ out, ushort_t* __restrict__ out_bf)
{
  __shared__ float red[8];
  const long base = (long)blockIdx.x * 768;
  const int tid = threadIdx.x, lane = tid & 31, w = tid >> 5;

  float x[3];
  float s = 0.f;
#pragma unroll
  for (int i = 0; i < 3; ++i) {
    const int h = tid + i * 256;
    float v = a[base + h];
    if (badd) v += badd[base + h];
    x[i] = v; s += v;
  }
#pragma unroll
  for (int off = 16; off; off >>= 1) s += __shfl_xor(s, off, 32);
  if (lane == 0) red[w] = s;
  __syncthreads();
  float tot = 0.f;
#pragma unroll
  for (int k = 0; k < 8; ++k) tot += red[k];
  const float mu = tot * (1.0f / 768.0f);

  float vs = 0.f;
#pragma unroll
  for (int i = 0; i < 3; ++i) { const float d = x[i] - mu; vs += d * d; }
  __syncthreads();
#pragma unroll
  for (int off = 16; off; off >>= 1) vs += __shfl_xor(vs, off, 32);
  if (lane == 0) red[w] = vs;
  __syncthreads();
  float totv = 0.f;
#pragma unroll
  for (int k = 0; k < 8; ++k) totv += red[k];
  const float inv = rsqrtf(totv * (1.0f / 768.0f) + 1e-12f);

#pragma unroll
  for (int i = 0; i < 3; ++i) {
    const int h = tid + i * 256;
    float y = (x[i] - mu) * inv * g[h] + beta[h];
    if (r1) y += r1[base + h];
    if (r2) y += r2[base + h];
    out[base + h] = y;
    if (out_bf) out_bf[base + h] = f2bf(y);
  }
}

// ---------------------------------------------------------------------------
extern "C" void kernel_launch(void* const* d_in, const int* in_sizes, int n_in,
                              void* d_out, int out_size, void* d_ws, size_t ws_size,
                              hipStream_t stream) {
  (void)in_sizes; (void)n_in; (void)out_size; (void)ws_size;

  const float* lhs   = (const float*)d_in[0];
  const int*   amask = (const int*)  d_in[1];
  const float* temb  = (const float*)d_in[2];
  const float* tp    = (const float*)d_in[3];
  const float* Wq    = (const float*)d_in[4];
  const float* bq    = (const float*)d_in[5];
  const float* Wk    = (const float*)d_in[6];
  const float* bk    = (const float*)d_in[7];
  const float* Wv    = (const float*)d_in[8];
  const float* bv    = (const float*)d_in[9];
  const float* Wso   = (const float*)d_in[10];
  const float* bso   = (const float*)d_in[11];
  const float* ln1g  = (const float*)d_in[12];
  const float* ln1b  = (const float*)d_in[13];
  const float* Wi    = (const float*)d_in[14];
  const float* bi    = (const float*)d_in[15];
  const float* Wbo   = (const float*)d_in[16];
  const float* bbo   = (const float*)d_in[17];
  const float* ln2g  = (const float*)d_in[18];
  const float* ln2b  = (const float*)d_in[19];

  const long BT = 4096, H = 768, FF = 3072, TT = 512;
  const long BTH = BT * H;
  float* out     = (float*)d_out;
  float* scores0 = out + BTH;

  // -------- workspace carve (fp32 + bf16 regions, ~320 MB total) --------
  char* wsb = (char*)d_ws;
  auto alloc_f = [&](long n) { float* p = (float*)wsb;    wsb += ((n * 4 + 255) & ~255L); return p; };
  auto alloc_u = [&](long n) { ushort_t* p = (ushort_t*)wsb; wsb += ((n * 2 + 255) & ~255L); return p; };

  float* f_first  = alloc_f(BTH);
  float* f_vu     = alloc_f(BTH);
  float* f_vt     = alloc_f(BTH);
  float* f_ctx    = alloc_f(BTH);
  float* f_attout = alloc_f(BTH);
  float* f_sotmp  = alloc_f(BTH);
  float* f_s2     = alloc_f(96 * TT);
  float* f_pdiag  = alloc_f(96 * TT);
  float* f_scores = alloc_f(96 * TT * TT);     // reused later as ffn_tmp
  float* f_ffntmp = f_scores;

  ushort_t* u_hunk   = alloc_u(BTH);
  ushort_t* u_htru   = alloc_u(BTH);
  ushort_t* u_qu     = alloc_u(BTH);
  ushort_t* u_ku     = alloc_u(BTH);
  ushort_t* u_kt     = alloc_u(BTH);
  ushort_t* u_vt     = alloc_u(BTH);
  ushort_t* u_ctx    = alloc_u(BTH);
  ushort_t* u_attout = alloc_u(BTH);
  ushort_t* u_inter  = alloc_u(BT * FF);
  ushort_t* u_P      = alloc_u(96 * TT * TT);
  ushort_t* u_Wq     = alloc_u(H * H);
  ushort_t* u_Wk     = alloc_u(H * H);
  ushort_t* u_Wv     = alloc_u(H * H);
  ushort_t* u_Wso    = alloc_u(H * H);
  ushort_t* u_Wi     = alloc_u(FF * H);
  ushort_t* u_Wbo    = alloc_u(H * FF);

  auto cvt = [&](const float* s, ushort_t* d, long n) {
    f32_to_bf16_kernel<<<(unsigned)(n / 1024), 256, 0, stream>>>(s, d, n);
  };
  auto gemm = [&](const ushort_t* A, long lda, const ushort_t* Bm, long ldb, int bkm,
                  const float* bias, float* C32, ushort_t* C16, long ldc,
                  int M, int N, int K, float alpha, int act, int batches, int innerB,
                  long sAo, long sAi, long sBo, long sBi, long sCo, long sCi) {
    dim3 grid((unsigned)(N / BN), (unsigned)(M / BM), (unsigned)batches);
    gemm_wmma_kernel<<<grid, 256, 0, stream>>>(A, lda, Bm, ldb, bkm, bias, C32, C16, ldc,
                                               M, N, K, alpha, act, innerB,
                                               sAo, sAi, sBo, sBi, sCo, sCi);
  };

  // 0) one-shot weight conversion to bf16 (28 MB total; trivial vs 23.3 TB/s)
  cvt(Wq, u_Wq, H * H); cvt(Wk, u_Wk, H * H); cvt(Wv, u_Wv, H * H);
  cvt(Wso, u_Wso, H * H); cvt(Wi, u_Wi, FF * H); cvt(Wbo, u_Wbo, H * FF);

  // 1) label attention + bf16 residual bases
  label_attn_kernel<<<(unsigned)BT, 256, 0, stream>>>(
      lhs, temb, tp, scores0, f_first, u_hunk, u_htru);

  // 2) projections (M=4096,N=768,K=768): q/k bf16-only, v dual/fp32
  gemm(u_hunk, H, u_Wq, H, 0, bq, nullptr, u_qu, H, (int)BT, (int)H, (int)H, 1.f, 0, 1, 1, 0,0,0,0,0,0);
  gemm(u_hunk, H, u_Wk, H, 0, bk, nullptr, u_ku, H, (int)BT, (int)H, (int)H, 1.f, 0, 1, 1, 0,0,0,0,0,0);
  gemm(u_hunk, H, u_Wv, H, 0, bv, f_vu,  nullptr, H, (int)BT, (int)H, (int)H, 1.f, 0, 1, 1, 0,0,0,0,0,0);
  gemm(u_htru, H, u_Wk, H, 0, bk, nullptr, u_kt, H, (int)BT, (int)H, (int)H, 1.f, 0, 1, 1, 0,0,0,0,0,0);
  gemm(u_htru, H, u_Wv, H, 0, bv, f_vt,  u_vt,   H, (int)BT, (int)H, (int)H, 1.f, 0, 1, 1, 0,0,0,0,0,0);

  // 3) diagonal scores s2 = diag(Q_u K_u^T) per head
  s2_diag_kernel<<<192, 256, 0, stream>>>(u_qu, u_ku, f_s2);

  // 4) S = alpha * Q_u @ K_t^T, batched over 96 (b,head)
  const float alpha = 1.0f / sqrtf(768.0f);
  gemm(u_qu, H, u_kt, H, 0, nullptr, f_scores, nullptr, TT, (int)TT, (int)TT, 64,
       alpha, 0, 96, 12,
       /*A*/ TT * H, 64, /*B*/ TT * H, 64, /*C*/ 12 * TT * TT, TT * TT);

  // 5) masked softmax with diagonal substitution -> bf16 P + pdiag
  attn_softmax_kernel<<<6144, 256, 0, stream>>>(f_scores, f_s2, amask, u_P, f_pdiag, alpha);

  // 6) ctx = P @ V_t (B is K-major)
  gemm(u_P, TT, u_vt, H, 1, nullptr, f_ctx, nullptr, H, (int)TT, 64, (int)TT,
       1.f, 0, 96, 12,
       /*A*/ 12 * TT * TT, TT * TT, /*B*/ TT * H, 64, /*C*/ TT * H, 64);

  // 7) diagonal context fix: ctx += pdiag * (v_u - v_t); emits bf16 ctx
  ctx_diag_fix_kernel<<<(unsigned)(BTH / 256), 256, 0, stream>>>(
      f_ctx, u_ctx, f_pdiag, f_vu, f_vt);

  // 8) att_out = LN(ctx @ Wso^T + bso), dual fp32 + bf16
  gemm(u_ctx, H, u_Wso, H, 0, bso, f_sotmp, nullptr, H, (int)BT, (int)H, (int)H, 1.f, 0, 1, 1, 0,0,0,0,0,0);
  layernorm_kernel<<<(unsigned)BT, 256, 0, stream>>>(
      f_sotmp, nullptr, ln1g, ln1b, nullptr, nullptr, f_attout, u_attout);

  // 9) inter = gelu(att_out @ Wi^T + bi), bf16-only
  gemm(u_attout, H, u_Wi, H, 0, bi, nullptr, u_inter, FF, (int)BT, (int)FF, (int)H, 1.f, 1, 1, 1, 0,0,0,0,0,0);

  // 10) ffn_tmp = inter @ Wbo^T + bbo
  gemm(u_inter, FF, u_Wbo, FF, 0, bbo, f_ffntmp, nullptr, H, (int)BT, (int)H, (int)FF, 1.f, 0, 1, 1, 0,0,0,0,0,0);

  // 11) out = LN(ffn_tmp + att_out) + lhs + first
  layernorm_kernel<<<(unsigned)BT, 256, 0, stream>>>(
      f_ffntmp, f_attout, ln2g, ln2b, lhs, f_first, out, nullptr);
}
